// LightweightMamba2_66116726555026
// MI455X (gfx1250) — compile-verified, matrix operand order checked by reading, not run on hardware
//
#include <hip/hip_runtime.h>
#include <hip/hip_bf16.h>
#include <stdint.h>

// ---------------------------------------------------------------------------
// Model dims
// ---------------------------------------------------------------------------
#define BATCH      512
#define SEQ        64
#define D_MODEL    512
#define D_INNER    1024
#define NHEADS     16
#define HEADDIM    64
#define D_STATE    128
#define CONV_DIM   1280          // D_INNER + 2*D_STATE
#define D_IN_PROJ  2336          // 2*D_INNER + 2*D_STATE + NHEADS + 16 unread cols
#define D_CONV     4
#define FC_HID     128
#define NUM_CLS    2
#define EPS        1e-5f
#define BS_ROWS    (BATCH * SEQ) // 32768

typedef __attribute__((ext_vector_type(16))) __bf16 v16bf;
typedef __attribute__((ext_vector_type(8)))  __bf16 v8bf;
typedef __attribute__((ext_vector_type(8)))  float  v8f;

// ---------------------------------------------------------------------------
// Fragment load helper: 16 bf16 = two contiguous 16B chunks
// (A-role: lane holds row m=lane&15; element j<8 -> K=k0+half*8+j,
//  j>=8 -> K=k0+16+half*8+(j-8).  B-role identical with n=lane&15.)
// ---------------------------------------------------------------------------
__device__ __forceinline__ v16bf frag_ld(const __bf16* p0, const __bf16* p1) {
  v8bf a = *(const v8bf*)p0;
  v8bf b = *(const v8bf*)p1;
  v16bf r;
#pragma unroll
  for (int i = 0; i < 8; ++i) { r[i] = a[i]; r[i + 8] = b[i]; }
  return r;
}

__device__ __forceinline__ v8f wmma_bf16(v16bf a, v16bf b, v8f c) {
  return __builtin_amdgcn_wmma_f32_16x16x32_bf16(
      /*neg_a=*/false, a, /*neg_b=*/false, b,
      /*c_mod=*/(short)0, c, /*reuse_a=*/false, /*reuse_b=*/false);
}

// 16B-per-lane async global->LDS copy (ASYNCcnt), GVS addressing:
//   lds[lds_off] = mem[saddr + voff]
__device__ __forceinline__ void async_g2l_b128(unsigned lds_off, unsigned voff,
                                               const void* sbase) {
  asm volatile("global_load_async_to_lds_b128 %0, %1, %2"
               :: "v"(lds_off), "v"(voff), "s"(sbase)
               : "memory");
}
__device__ __forceinline__ void wait_async0() {
  asm volatile("s_wait_asynccnt 0x0" ::: "memory");
}

// ---------------------------------------------------------------------------
// GEMM:  C[M,N] = A[M,K] * W[N,K]^T   (bf16 in, f32 out)
// grid = (M/128, ceil(N/64)), block = 128 (4 waves).
// Each wave computes 32(M) x 64(N): 2 A-frags reused over 4 B-frags
// -> 8 WMMAs per 6 fragment loads per k-step.
// N need not be a multiple of 64 (reads clamped, writes guarded).
// ---------------------------------------------------------------------------
__global__ void wmma_gemm_nt(const __bf16* __restrict__ A,
                             const __bf16* __restrict__ W,
                             float* __restrict__ C,
                             int M, int N, int K) {
  const int wave = threadIdx.x >> 5;
  const int lane = threadIdx.x & 31;
  const int col  = lane & 15;
  const int half = lane >> 4;
  const int m_base = blockIdx.x * 128 + wave * 32;
  const int n_base = blockIdx.y * 64;

  v8f acc[2][4] = {};
  const __bf16* arow0 = A + (size_t)(m_base + col) * K;
  const __bf16* arow1 = A + (size_t)(m_base + 16 + col) * K;

  const __bf16* wrow[4];
#pragma unroll
  for (int nt = 0; nt < 4; ++nt) {
    int n = n_base + nt * 16 + col;
    if (n >= N) n = N - 1;
    wrow[nt] = W + (size_t)n * K;
  }

  for (int k0 = 0; k0 < K; k0 += 32) {
    if (k0 + 32 < K) {
      __builtin_prefetch(arow0 + k0 + 32, 0, 1); // global_prefetch_b8
      __builtin_prefetch(arow1 + k0 + 32, 0, 1);
    }
    v16bf a0 = frag_ld(arow0 + k0 + half * 8, arow0 + k0 + 16 + half * 8);
    v16bf a1 = frag_ld(arow1 + k0 + half * 8, arow1 + k0 + 16 + half * 8);
#pragma unroll
    for (int nt = 0; nt < 4; ++nt) {
      v16bf bfg = frag_ld(wrow[nt] + k0 + half * 8, wrow[nt] + k0 + 16 + half * 8);
      acc[0][nt] = wmma_bf16(a0, bfg, acc[0][nt]);
      acc[1][nt] = wmma_bf16(a1, bfg, acc[1][nt]);
    }
  }

#pragma unroll
  for (int mt = 0; mt < 2; ++mt) {
#pragma unroll
    for (int nt = 0; nt < 4; ++nt) {
      int n = n_base + nt * 16 + col;
      if (n < N) {
#pragma unroll
        for (int r = 0; r < 8; ++r) {
          int m = m_base + mt * 16 + r + 8 * half;
          C[(size_t)m * N + n] = acc[mt][nt][r];
        }
      }
    }
  }
}

// ---------------------------------------------------------------------------
// scores[b] = Cm[b](64x128) @ Bm[b](64x128)^T via WMMA, operands staged in
// LDS with async global->LDS b128 copies.  grid = BATCH, block = 128.
// ---------------------------------------------------------------------------
__global__ void scores_kernel(const __bf16* __restrict__ Cmb,
                              const __bf16* __restrict__ Bmb,
                              float* __restrict__ scores) {
  __shared__ __align__(16) __bf16 Cs[SEQ * D_STATE]; // 16 KB
  __shared__ __align__(16) __bf16 Bs[SEQ * D_STATE]; // 16 KB
  const int b = blockIdx.x;
  const int tid = threadIdx.x;
  const __bf16* cg = Cmb + (size_t)b * SEQ * D_STATE;
  const __bf16* bg = Bmb + (size_t)b * SEQ * D_STATE;
  const unsigned cs_base = (unsigned)(uintptr_t)(void*)&Cs[0];
  const unsigned bs_base = (unsigned)(uintptr_t)(void*)&Bs[0];

  // 2 x 16KB / (128 threads * 16B) = 8 async b128 copies per thread per tile
#pragma unroll
  for (int i = 0; i < 8; ++i) {
    unsigned off = (unsigned)(tid + i * 128) * 16u;
    async_g2l_b128(cs_base + off, off, cg);
    async_g2l_b128(bs_base + off, off, bg);
  }
  wait_async0();
  __syncthreads();

  const int wave = tid >> 5, lane = tid & 31;
  const int col = lane & 15, half = lane >> 4;
  const int mrow = wave * 16 + col;
  v8f acc[4] = {};
  for (int k0 = 0; k0 < D_STATE; k0 += 32) {
    v16bf af = frag_ld(&Cs[mrow * D_STATE + k0 + half * 8],
                       &Cs[mrow * D_STATE + k0 + 16 + half * 8]);
#pragma unroll
    for (int nt = 0; nt < 4; ++nt) {
      int n = nt * 16 + col;
      v16bf bfg = frag_ld(&Bs[n * D_STATE + k0 + half * 8],
                          &Bs[n * D_STATE + k0 + 16 + half * 8]);
      acc[nt] = wmma_bf16(af, bfg, acc[nt]);
    }
  }
#pragma unroll
  for (int nt = 0; nt < 4; ++nt) {
    int n = nt * 16 + col;
#pragma unroll
    for (int r = 0; r < 8; ++r) {
      int m = wave * 16 + r + 8 * half;
      scores[((size_t)b * SEQ + m) * SEQ + n] = acc[nt][r];
    }
  }
}

// ---------------------------------------------------------------------------
// Embedding + positional:  h[bs,d] = emb[x[bs],d] + pos[s,d]; also bf16 copy
// ---------------------------------------------------------------------------
__global__ void embed_kernel(const int* __restrict__ x,
                             const float* __restrict__ emb,
                             const float* __restrict__ pos,
                             float* __restrict__ h, __bf16* __restrict__ hbf) {
  int bs = blockIdx.x;
  int s = bs % SEQ;
  int tok = x[bs];
  for (int d = threadIdx.x; d < D_MODEL; d += blockDim.x) {
    float v = emb[(size_t)tok * D_MODEL + d] + pos[(size_t)s * D_MODEL + d];
    h[(size_t)bs * D_MODEL + d] = v;
    hbf[(size_t)bs * D_MODEL + d] = (__bf16)v;
  }
}

__global__ void f32_to_bf16_kernel(const float* __restrict__ src,
                                   __bf16* __restrict__ dst, long long n) {
  long long i = (long long)blockIdx.x * blockDim.x + threadIdx.x;
  if (i < n) dst[i] = (__bf16)src[i];
}

// ---------------------------------------------------------------------------
// Depthwise causal conv (D_CONV=4) + SiLU over the xBC slice of zxbcdt,
// also emit bf16 B/C matrices for the scores WMMA. grid = BS_ROWS, block 256
// ---------------------------------------------------------------------------
__global__ void conv_silu_kernel(const float* __restrict__ zx,
                                 const float* __restrict__ conv_w,
                                 const float* __restrict__ conv_b,
                                 float* __restrict__ xconv,
                                 __bf16* __restrict__ Bmb,
                                 __bf16* __restrict__ Cmb) {
  int bs = blockIdx.x;
  int b = bs / SEQ, s = bs % SEQ;
  for (int c = threadIdx.x; c < CONV_DIM; c += blockDim.x) {
    float acc = conv_b[c];
#pragma unroll
    for (int t = 0; t < D_CONV; ++t) {
      int sj = s - (D_CONV - 1) + t;
      if (sj >= 0)
        acc += zx[((size_t)b * SEQ + sj) * D_IN_PROJ + D_INNER + c] * conv_w[c * D_CONV + t];
    }
    float v = acc / (1.f + __expf(-acc)); // silu
    xconv[(size_t)bs * CONV_DIM + c] = v;
    if (c >= D_INNER && c < D_INNER + D_STATE)
      Bmb[(size_t)bs * D_STATE + (c - D_INNER)] = (__bf16)v;
    else if (c >= D_INNER + D_STATE)
      Cmb[(size_t)bs * D_STATE + (c - D_INNER - D_STATE)] = (__bf16)v;
  }
}

// ---------------------------------------------------------------------------
// dt = softplus(dt_raw + bias); cum = cumsum(dt * A) along seq.
// one thread per (b,h): 8192 threads total.
// ---------------------------------------------------------------------------
__global__ void dt_cumsum_kernel(const float* __restrict__ zx,
                                 const float* __restrict__ dt_bias,
                                 const float* __restrict__ A_log,
                                 float* __restrict__ dt, float* __restrict__ cum) {
  int idx = blockIdx.x * blockDim.x + threadIdx.x;
  if (idx >= BATCH * NHEADS) return;
  int b = idx / NHEADS, h = idx % NHEADS;
  float A = -__expf(A_log[h]);
  float bias = dt_bias[h];
  float c = 0.f;
  for (int s = 0; s < SEQ; ++s) {
    float raw = zx[((size_t)b * SEQ + s) * D_IN_PROJ + D_INNER + CONV_DIM + h] + bias;
    float d = (raw > 20.f) ? raw : log1pf(__expf(raw));
    c += d * A;
    dt[((size_t)b * SEQ + s) * NHEADS + h] = d;
    cum[((size_t)b * SEQ + s) * NHEADS + h] = c;
  }
}

// ---------------------------------------------------------------------------
// SSM "attention" per (b,h): G[i,j] = scores[i,j]*exp(cum_i - cum_j) (causal),
// V[j,p] = dt_j * xs[j,p];  y = G(64x64) @ V(64x64) via WMMA from LDS.
// grid = (BATCH, NHEADS), block = 128 (4 waves).
// ---------------------------------------------------------------------------
__global__ void ssm_attn_kernel(const float* __restrict__ scores,
                                const float* __restrict__ cum,
                                const float* __restrict__ dt,
                                const float* __restrict__ xconv,
                                float* __restrict__ yattn) {
  __shared__ __align__(16) __bf16 Gs[SEQ][72];   // 72 keeps 16B align + bank spread
  __shared__ __align__(16) __bf16 Vt[HEADDIM][72];
  __shared__ float  cums[SEQ], dts[SEQ];
  const int b = blockIdx.x, h = blockIdx.y;
  const int tid = threadIdx.x;

  if (tid < SEQ) {
    cums[tid] = cum[((size_t)b * SEQ + tid) * NHEADS + h];
    dts[tid]  = dt[((size_t)b * SEQ + tid) * NHEADS + h];
  }
  __syncthreads();

  for (int idx = tid; idx < SEQ * SEQ; idx += 128) {
    int i = idx >> 6, j = idx & 63;
    float g = 0.f;
    if (j <= i)
      g = scores[((size_t)b * SEQ + i) * SEQ + j] * __expf(cums[i] - cums[j]);
    Gs[i][j] = (__bf16)g;
  }
  for (int idx = tid; idx < HEADDIM * SEQ; idx += 128) {
    int p = idx >> 6, j = idx & 63; // Vt[p][j] = dt[j]*xs[j,p]
    float v = dts[j] * xconv[((size_t)b * SEQ + j) * CONV_DIM + h * HEADDIM + p];
    Vt[p][j] = (__bf16)v;
  }
  __syncthreads();

  const int wave = tid >> 5, lane = tid & 31;
  const int col = lane & 15, half = lane >> 4;
  const int mrow = wave * 16 + col;
  v8f acc[4] = {};
  for (int k0 = 0; k0 < SEQ; k0 += 32) {
    v16bf af = frag_ld(&Gs[mrow][k0 + half * 8], &Gs[mrow][k0 + 16 + half * 8]);
#pragma unroll
    for (int nt = 0; nt < 4; ++nt) {
      int n = nt * 16 + col;
      v16bf bfg = frag_ld(&Vt[n][k0 + half * 8], &Vt[n][k0 + 16 + half * 8]);
      acc[nt] = wmma_bf16(af, bfg, acc[nt]);
    }
  }
#pragma unroll
  for (int nt = 0; nt < 4; ++nt) {
    int p = nt * 16 + col;
#pragma unroll
    for (int r = 0; r < 8; ++r) {
      int i = wave * 16 + r + 8 * half;
      yattn[((size_t)b * SEQ + i) * D_INNER + h * HEADDIM + p] = acc[nt][r];
    }
  }
}

// ---------------------------------------------------------------------------
// y = (yattn + D_skip*xs) * silu(z); RMSNorm over D_INNER; emit bf16 for
// out_proj.  grid = BS_ROWS, block = 256 (4 elems/thread).
// ---------------------------------------------------------------------------
__global__ void gate_rmsnorm_kernel(const float* __restrict__ yattn,
                                    const float* __restrict__ xconv,
                                    const float* __restrict__ zx,
                                    const float* __restrict__ D_skip,
                                    const float* __restrict__ norm_w,
                                    __bf16* __restrict__ ybf) {
  __shared__ float red[256];
  const int bs = blockIdx.x;
  float vals[4];
  float ss = 0.f;
#pragma unroll
  for (int i = 0; i < 4; ++i) {
    int d = threadIdx.x + i * 256;
    int h = d >> 6;
    float y = yattn[(size_t)bs * D_INNER + d] + D_skip[h] * xconv[(size_t)bs * CONV_DIM + d];
    float z = zx[(size_t)bs * D_IN_PROJ + d];
    y *= z / (1.f + __expf(-z));
    vals[i] = y;
    ss += y * y;
  }
  red[threadIdx.x] = ss;
  __syncthreads();
  for (int st = 128; st > 0; st >>= 1) {
    if (threadIdx.x < st) red[threadIdx.x] += red[threadIdx.x + st];
    __syncthreads();
  }
  float inv = rsqrtf(red[0] / (float)D_INNER + EPS);
#pragma unroll
  for (int i = 0; i < 4; ++i) {
    int d = threadIdx.x + i * 256;
    ybf[(size_t)bs * D_INNER + d] = (__bf16)(vals[i] * inv * norm_w[d]);
  }
}

// ---------------------------------------------------------------------------
// h += LayerNorm(yproj); refresh bf16 copy.  grid = BS_ROWS, block = 256.
// ---------------------------------------------------------------------------
__global__ void residual_ln_kernel(float* __restrict__ h,
                                   const float* __restrict__ yproj,
                                   const float* __restrict__ ln_g,
                                   const float* __restrict__ ln_b,
                                   __bf16* __restrict__ hbf) {
  __shared__ float red[256];
  __shared__ float mu_s, var_s;
  const int bs = blockIdx.x;
  float v[2];
  float s = 0.f;
#pragma unroll
  for (int i = 0; i < 2; ++i) {
    int d = threadIdx.x + i * 256;
    v[i] = yproj[(size_t)bs * D_MODEL + d];
    s += v[i];
  }
  red[threadIdx.x] = s;
  __syncthreads();
  for (int st = 128; st > 0; st >>= 1) {
    if (threadIdx.x < st) red[threadIdx.x] += red[threadIdx.x + st];
    __syncthreads();
  }
  if (threadIdx.x == 0) mu_s = red[0] / (float)D_MODEL;
  __syncthreads();
  float mu = mu_s;
  s = 0.f;
#pragma unroll
  for (int i = 0; i < 2; ++i) { float t = v[i] - mu; s += t * t; }
  red[threadIdx.x] = s;
  __syncthreads();
  for (int st = 128; st > 0; st >>= 1) {
    if (threadIdx.x < st) red[threadIdx.x] += red[threadIdx.x + st];
    __syncthreads();
  }
  if (threadIdx.x == 0) var_s = red[0] / (float)D_MODEL;
  __syncthreads();
  float inv = rsqrtf(var_s + EPS);
#pragma unroll
  for (int i = 0; i < 2; ++i) {
    int d = threadIdx.x + i * 256;
    float out = h[(size_t)bs * D_MODEL + d] + (v[i] - mu) * inv * ln_g[d] + ln_b[d];
    h[(size_t)bs * D_MODEL + d] = out;
    hbf[(size_t)bs * D_MODEL + d] = (__bf16)out;
  }
}

// ---------------------------------------------------------------------------
// Masked mean-pool over seq.  grid = BATCH, block = 256.
// ---------------------------------------------------------------------------
__global__ void pool_kernel(const float* __restrict__ h, float* __restrict__ pooled) {
  __shared__ float red[256];
  __shared__ float rowsum[SEQ];
  __shared__ float cnt_s;
  const int b = blockIdx.x;
  const int row = threadIdx.x >> 2, part = threadIdx.x & 3;
  float s = 0.f;
  for (int d = part * 128; d < part * 128 + 128; ++d)
    s += h[((size_t)b * SEQ + row) * D_MODEL + d];
  red[threadIdx.x] = s;
  __syncthreads();
  if (part == 0)
    rowsum[row] = red[threadIdx.x] + red[threadIdx.x + 1] + red[threadIdx.x + 2] + red[threadIdx.x + 3];
  __syncthreads();
  if (threadIdx.x == 0) {
    float c = 0.f;
    for (int s2 = 0; s2 < SEQ; ++s2) c += (rowsum[s2] != 0.f) ? 1.f : 0.f;
    cnt_s = fmaxf(c, 1.f);
  }
  __syncthreads();
  for (int d = threadIdx.x; d < D_MODEL; d += 256) {
    float acc = 0.f;
    for (int s2 = 0; s2 < SEQ; ++s2)
      if (rowsum[s2] != 0.f) acc += h[((size_t)b * SEQ + s2) * D_MODEL + d];
    pooled[(size_t)b * D_MODEL + d] = acc / cnt_s;
  }
}

// ---------------------------------------------------------------------------
// FC head: hid = relu(pooled @ fc1^T + b1); out = hid @ fc2^T + b2
// grid = BATCH, block = 128
// ---------------------------------------------------------------------------
__global__ void fc_kernel(const float* __restrict__ pooled,
                          const float* __restrict__ w1, const float* __restrict__ b1,
                          const float* __restrict__ w2, const float* __restrict__ b2,
                          float* __restrict__ out) {
  __shared__ float hid[FC_HID];
  const int b = blockIdx.x;
  float acc = b1[threadIdx.x];
  for (int d = 0; d < D_MODEL; ++d)
    acc += pooled[(size_t)b * D_MODEL + d] * w1[(size_t)threadIdx.x * D_MODEL + d];
  hid[threadIdx.x] = fmaxf(acc, 0.f);
  __syncthreads();
  if (threadIdx.x < NUM_CLS) {
    float o = b2[threadIdx.x];
    for (int j = 0; j < FC_HID; ++j) o += hid[j] * w2[threadIdx.x * FC_HID + j];
    out[b * NUM_CLS + threadIdx.x] = o;
  }
}

// ---------------------------------------------------------------------------
// Orchestration
// ---------------------------------------------------------------------------
extern "C" void kernel_launch(void* const* d_in, const int* in_sizes, int n_in,
                              void* d_out, int out_size, void* d_ws, size_t ws_size,
                              hipStream_t stream) {
  const int*   x         = (const int*)  d_in[0];
  const float* emb       = (const float*)d_in[1];
  const float* pos       = (const float*)d_in[2];
  const float* in_proj_w = (const float*)d_in[3];
  const float* conv_w    = (const float*)d_in[4];
  const float* conv_b    = (const float*)d_in[5];
  const float* dt_bias   = (const float*)d_in[6];
  const float* A_log     = (const float*)d_in[7];
  const float* D_skip    = (const float*)d_in[8];
  const float* norm_w    = (const float*)d_in[9];
  const float* out_projw = (const float*)d_in[10];
  const float* ln_g      = (const float*)d_in[11];
  const float* ln_b      = (const float*)d_in[12];
  const float* fc1_w     = (const float*)d_in[13];
  const float* fc1_b     = (const float*)d_in[14];
  const float* fc2_w     = (const float*)d_in[15];
  const float* fc2_b     = (const float*)d_in[16];
  float* out = (float*)d_out;

  // workspace arena
  char* ws = (char*)d_ws;
  size_t off = 0;
  auto alloc = [&](size_t bytes) -> void* {
    void* p = ws + off;
    off = (off + bytes + 255) & ~(size_t)255;
    return p;
  };
  float*  h      = (float*) alloc((size_t)BS_ROWS * D_MODEL * 4);
  __bf16* hbf    = (__bf16*)alloc((size_t)BS_ROWS * D_MODEL * 2);
  float*  zx     = (float*) alloc((size_t)BS_ROWS * D_IN_PROJ * 4);
  float*  xconv  = (float*) alloc((size_t)BS_ROWS * CONV_DIM * 4);
  __bf16* Bmb    = (__bf16*)alloc((size_t)BS_ROWS * D_STATE * 2);
  __bf16* Cmb    = (__bf16*)alloc((size_t)BS_ROWS * D_STATE * 2);
  float*  dt     = (float*) alloc((size_t)BS_ROWS * NHEADS * 4);
  float*  cum    = (float*) alloc((size_t)BS_ROWS * NHEADS * 4);
  float*  scores = (float*) alloc((size_t)BS_ROWS * SEQ * 4);
  float*  yattn  = (float*) alloc((size_t)BS_ROWS * D_INNER * 4);
  __bf16* ybf    = (__bf16*)alloc((size_t)BS_ROWS * D_INNER * 2);
  float*  yproj  = (float*) alloc((size_t)BS_ROWS * D_MODEL * 4);
  __bf16* wbf    = (__bf16*)alloc((size_t)D_IN_PROJ * D_MODEL * 2);
  float*  pooled = (float*) alloc((size_t)BATCH * D_MODEL * 4);

  embed_kernel<<<BS_ROWS, 256, 0, stream>>>(x, emb, pos, h, hbf);

  for (int l = 0; l < 2; ++l) {
    // in_proj weights -> bf16
    long long nw = (long long)D_IN_PROJ * D_MODEL;
    f32_to_bf16_kernel<<<(unsigned)((nw + 255) / 256), 256, 0, stream>>>(
        in_proj_w + (size_t)l * nw, wbf, nw);
    // zx = h @ in_proj^T   (M=32768, N=2336, K=512)
    wmma_gemm_nt<<<dim3(BS_ROWS / 128, (D_IN_PROJ + 63) / 64), 128, 0, stream>>>(
        hbf, wbf, zx, BS_ROWS, D_IN_PROJ, D_MODEL);

    conv_silu_kernel<<<BS_ROWS, 256, 0, stream>>>(
        zx, conv_w + (size_t)l * CONV_DIM * D_CONV, conv_b + (size_t)l * CONV_DIM,
        xconv, Bmb, Cmb);

    dt_cumsum_kernel<<<(BATCH * NHEADS + 255) / 256, 256, 0, stream>>>(
        zx, dt_bias + l * NHEADS, A_log + l * NHEADS, dt, cum);

    // scores[b] = C(64x128) @ B(64x128)^T, async-LDS staged, batched over 512
    scores_kernel<<<BATCH, 128, 0, stream>>>(Cmb, Bmb, scores);

    ssm_attn_kernel<<<dim3(BATCH, NHEADS), 128, 0, stream>>>(
        scores, cum, dt, xconv, yattn);

    gate_rmsnorm_kernel<<<BS_ROWS, 256, 0, stream>>>(
        yattn, xconv, zx, D_skip + l * NHEADS, norm_w + (size_t)l * D_INNER, ybf);

    // out_proj weights -> bf16
    long long nw2 = (long long)D_MODEL * D_INNER;
    f32_to_bf16_kernel<<<(unsigned)((nw2 + 255) / 256), 256, 0, stream>>>(
        out_projw + (size_t)l * nw2, wbf, nw2);
    // yproj = y @ out_proj^T   (M=32768, N=512, K=1024)
    wmma_gemm_nt<<<dim3(BS_ROWS / 128, D_MODEL / 64), 128, 0, stream>>>(
        ybf, wbf, yproj, BS_ROWS, D_MODEL, D_INNER);

    residual_ln_kernel<<<BS_ROWS, 256, 0, stream>>>(
        h, yproj, ln_g + (size_t)l * D_MODEL, ln_b + (size_t)l * D_MODEL, hbf);
  }

  pool_kernel<<<BATCH, 256, 0, stream>>>(h, pooled);
  fc_kernel<<<BATCH, FC_HID, 0, stream>>>(pooled, fc1_w, fc1_b, fc2_w, fc2_b, out);
}
